// attentive_node_features_20383914787443
// MI455X (gfx1250) — matrix operand order repaired and checked by reading, not compiled, not run on hardware
//
#include <hip/hip_runtime.h>
#include <hip/hip_bf16.h>

// Problem constants (match reference: B=16, N=2048, D=512)
#define B_   16
#define N_   2048
#define D_   512
#define NEGC (-1e9f)

#define XS_LD (D_ + 8)   // LDS stride (bf16 elems) for x tile, 16B-aligned, bank-skewed
#define PS_LD (N_ + 8)   // LDS stride (bf16 elems) for prob tile

typedef __attribute__((ext_vector_type(16))) __bf16 v16bf_t;
typedef __attribute__((ext_vector_type(8)))  float  v8f_t;

union BV16 { v16bf_t v; uint4 q[2]; };

// ---- WMMA operand loaders (wave32, 16x16x32 bf16) -------------------------
// A operand (16xK32): lane L (<16) holds row M=L, K={0..7,16..23}; lane L+16
// holds row M=L, K={8..15,24..31}.  Source is row-major MxK with leading dim ld.
__device__ __forceinline__ v16bf_t load_a32(const __bf16* base, int ld, int lane) {
  const int m  = lane & 15;
  const int hi = lane >> 4;
  const __bf16* p = base + (size_t)m * ld + hi * 8;
  BV16 u;
  u.q[0] = *reinterpret_cast<const uint4*>(p);       // K 0..7   / 8..15
  u.q[1] = *reinterpret_cast<const uint4*>(p + 16);  // K 16..23 / 24..31
  return u.v;
}

// B operand (K32x16): lane j (<16) holds column N=j, K={0..7,16..23}; lane
// j+16 holds K={8..15,24..31}.  Source is B-transposed row-major (N x K) with ld.
__device__ __forceinline__ v16bf_t load_bt32(const __bf16* base, int ld, int lane) {
  const int n  = lane & 15;
  const int hi = lane >> 4;
  const __bf16* p = base + (size_t)n * ld + hi * 8;
  BV16 u;
  u.q[0] = *reinterpret_cast<const uint4*>(p);       // K 0..7   / 8..15
  u.q[1] = *reinterpret_cast<const uint4*>(p + 16);  // K 16..23 / 24..31
  return u.v;
}

#define WMMA_BF16(a, b, c) \
  __builtin_amdgcn_wmma_f32_16x16x32_bf16(false, (a), false, (b), (short)0, (c), false, false)

// ---- Prep kernels: f32 -> bf16 copies ------------------------------------
__global__ void cvt_lin_kernel(const float* __restrict__ src, __bf16* __restrict__ dst,
                               size_t n) {
  size_t i = (size_t)blockIdx.x * blockDim.x + threadIdx.x;
  size_t stride = (size_t)gridDim.x * blockDim.x;
  for (; i < n; i += stride) dst[i] = (__bf16)src[i];
}

// fdm[b][d][m] = features[b][m][d]  (coalesced writes)
__global__ void cvt_tr_kernel(const float* __restrict__ feat, __bf16* __restrict__ fdm) {
  size_t total = (size_t)B_ * N_ * D_;
  size_t i = (size_t)blockIdx.x * blockDim.x + threadIdx.x;
  size_t stride = (size_t)gridDim.x * blockDim.x;
  for (; i < total; i += stride) {
    int m = (int)(i % N_);
    int d = (int)((i / N_) % D_);
    int b = (int)(i / ((size_t)N_ * D_));
    fdm[i] = (__bf16)feat[(size_t)b * N_ * D_ + (size_t)m * D_ + d];
  }
}

// ---- Fused attention kernel ----------------------------------------------
// One workgroup (128 threads = 4 wave32) per (batch, 16-row query tile).
// All GEMM phases: 4-way output-tile register blocking + manual 2x-unrolled
// ping-pong double buffering of the streamed operand (no rotation copies).
__global__ __launch_bounds__(128)
void fused_attn_kernel(const int*    __restrict__ lengths,
                       const float*  __restrict__ bias,
                       const __bf16* __restrict__ fmd,   // [B][N][D] bf16
                       const __bf16* __restrict__ fdm,   // [B][D][N] bf16
                       const __bf16* __restrict__ wbf,   // [D][D] bf16 (row e, contiguous d)
                       float*        __restrict__ out)   // [B][N][D] f32
{
  __shared__ __bf16 xs[16 * XS_LD];    // x tile: 16 rows x D, bf16
  __shared__ __bf16 ps[16 * PS_LD];    // exp(tanh(scores)): 16 rows x N, bf16
  __shared__ float  rowsum[16];

  const int b    = blockIdx.x;
  const int n0   = blockIdx.y * 16;
  const int tid  = threadIdx.x;
  const int lane = tid & 31;
  const int wave = tid >> 5;
  const int hi   = lane >> 4;
  const int lo   = lane & 15;

  const __bf16* fb  = fmd + (size_t)b * N_ * D_;
  const __bf16* ftb = fdm + (size_t)b * D_ * N_;

  if (tid < 16) rowsum[tid] = 0.0f;

  // ---- Phase 1: x_tile = f[n0:n0+16,:] @ W^T + bias  -> LDS (bf16) -------
  for (int eg = wave; eg < (D_ / 16) / 4; eg += 4) {
    const int e0 = eg * 64;
    const __bf16* abase = fb + (size_t)n0 * D_;
    const __bf16* bbase = wbf + (size_t)e0 * D_;
    v8f_t acc[4] = {{}, {}, {}, {}};
    v16bf_t a0 = load_a32(abase, D_, lane), a1;
    v16bf_t b0[4], b1[4];
#pragma unroll
    for (int j = 0; j < 4; ++j) b0[j] = load_bt32(bbase + (size_t)(16 * j) * D_, D_, lane);
#pragma unroll 1
    for (int k0 = 0; k0 < D_; k0 += 64) {
      // half A: prefetch k0+32 into buffer 1, consume buffer 0
      a1 = load_a32(abase + k0 + 32, D_, lane);
#pragma unroll
      for (int j = 0; j < 4; ++j)
        b1[j] = load_bt32(bbase + (size_t)(16 * j) * D_ + k0 + 32, D_, lane);
#pragma unroll
      for (int j = 0; j < 4; ++j) acc[j] = WMMA_BF16(a0, b0[j], acc[j]);
      // half B: prefetch k0+64 (wrap) into buffer 0, consume buffer 1
      const int kn = (k0 + 64 < D_) ? (k0 + 64) : 0;
      a0 = load_a32(abase + kn, D_, lane);
#pragma unroll
      for (int j = 0; j < 4; ++j)
        b0[j] = load_bt32(bbase + (size_t)(16 * j) * D_ + kn, D_, lane);
#pragma unroll
      for (int j = 0; j < 4; ++j) acc[j] = WMMA_BF16(a1, b1[j], acc[j]);
    }
#pragma unroll
    for (int j = 0; j < 4; ++j) {
      const int ej = e0 + 16 * j + lo;
      const float bval = bias[ej];
#pragma unroll
      for (int r = 0; r < 8; ++r) {
        const int nl = r + (hi << 3);          // C layout: VGPR r -> row r / r+8
        xs[nl * XS_LD + ej] = (__bf16)(acc[j][r] + bval);
      }
    }
  }
  __syncthreads();

  // ---- Phase 2: scores, mask -> tanh -> exp, row sums --------------------
  const int len = lengths[b];
  for (int mg = wave; mg < (N_ / 16) / 4; mg += 4) {
    const int m0 = mg * 64;
    const __bf16* bbase = fb + (size_t)m0 * D_;
    v8f_t acc[4] = {{}, {}, {}, {}};
    v16bf_t b0[4], b1[4];
#pragma unroll
    for (int j = 0; j < 4; ++j) b0[j] = load_bt32(bbase + (size_t)(16 * j) * D_, D_, lane);
#pragma unroll 1
    for (int k0 = 0; k0 < D_; k0 += 64) {
#pragma unroll
      for (int j = 0; j < 4; ++j)
        b1[j] = load_bt32(bbase + (size_t)(16 * j) * D_ + k0 + 32, D_, lane);
      v16bf_t av = load_a32(xs + k0, XS_LD, lane);         // A from LDS (queries)
#pragma unroll
      for (int j = 0; j < 4; ++j) acc[j] = WMMA_BF16(av, b0[j], acc[j]);
      const int kn = (k0 + 64 < D_) ? (k0 + 64) : 0;
#pragma unroll
      for (int j = 0; j < 4; ++j)
        b0[j] = load_bt32(bbase + (size_t)(16 * j) * D_ + kn, D_, lane);
      av = load_a32(xs + k0 + 32, XS_LD, lane);
#pragma unroll
      for (int j = 0; j < 4; ++j) acc[j] = WMMA_BF16(av, b1[j], acc[j]);
    }
#pragma unroll
    for (int j = 0; j < 4; ++j) {
      const int mgl = m0 + 16 * j + lo;        // key index (C-layout column)
#pragma unroll
      for (int r = 0; r < 8; ++r) {
        const int nl = r + (hi << 3);
        const int ng = n0 + nl;
        const bool keep = (mgl < len) && (ng >= mgl);   // padding & causal
        float s = keep ? acc[j][r] : NEGC;
        float p = expf(tanhf(s));              // tanh bounds => stable softmax
        ps[nl * PS_LD + mgl] = (__bf16)p;
        // reduce over the 16 lanes of this half-wave (same row, distinct keys)
        p += __shfl_xor(p, 1, 32);
        p += __shfl_xor(p, 2, 32);
        p += __shfl_xor(p, 4, 32);
        p += __shfl_xor(p, 8, 32);
        if (lo == 0) atomicAdd(&rowsum[nl], p);
      }
    }
  }
  __syncthreads();

  // ---- Phase 3: out = (p / rowsum) @ f  ----------------------------------
  float inv[8];
#pragma unroll
  for (int r = 0; r < 8; ++r) inv[r] = 1.0f / rowsum[r + (hi << 3)];

  for (int dg = wave; dg < (D_ / 16) / 4; dg += 4) {
    const int d0 = dg * 64;
    const __bf16* bbase = ftb + (size_t)d0 * N_;
    v8f_t acc[4] = {{}, {}, {}, {}};
    v16bf_t b0[4], b1[4];
#pragma unroll
    for (int j = 0; j < 4; ++j) b0[j] = load_bt32(bbase + (size_t)(16 * j) * N_, N_, lane);
#pragma unroll 1
    for (int m0 = 0; m0 < N_; m0 += 64) {
#pragma unroll
      for (int j = 0; j < 4; ++j)
        b1[j] = load_bt32(bbase + (size_t)(16 * j) * N_ + m0 + 32, N_, lane);
      v16bf_t av = load_a32(ps + m0, PS_LD, lane);         // A from LDS (probs)
#pragma unroll
      for (int j = 0; j < 4; ++j) acc[j] = WMMA_BF16(av, b0[j], acc[j]);
      const int mn = (m0 + 64 < N_) ? (m0 + 64) : 0;
#pragma unroll
      for (int j = 0; j < 4; ++j)
        b0[j] = load_bt32(bbase + (size_t)(16 * j) * N_ + mn, N_, lane);
      av = load_a32(ps + m0 + 32, PS_LD, lane);
#pragma unroll
      for (int j = 0; j < 4; ++j) acc[j] = WMMA_BF16(av, b1[j], acc[j]);
    }
#pragma unroll
    for (int j = 0; j < 4; ++j) {
#pragma unroll
      for (int r = 0; r < 8; ++r) {
        const int nl = r + (hi << 3);
        const float o = acc[j][r] * inv[r];
        out[((size_t)b * N_ + n0 + nl) * D_ + d0 + 16 * j + lo] = o;
      }
    }
  }
}

// ---- Host launch ----------------------------------------------------------
extern "C" void kernel_launch(void* const* d_in, const int* in_sizes, int n_in,
                              void* d_out, int out_size, void* d_ws, size_t ws_size,
                              hipStream_t stream) {
  const float* features = (const float*)d_in[0];   // (B,N,D) f32
  const int*   lengths  = (const int*)d_in[1];     // (B,)    i32
  const float* W        = (const float*)d_in[2];   // (D,D)   f32
  const float* bias     = (const float*)d_in[3];   // (D,)    f32
  float*       out      = (float*)d_out;

  const size_t nfeat = (size_t)B_ * N_ * D_;       // 16,777,216
  __bf16* fmd = (__bf16*)d_ws;                     // [B][N][D]
  __bf16* fdm = fmd + nfeat;                       // [B][D][N]
  __bf16* wbf = fdm + nfeat;                       // [D][D]

  // Prep: bf16 copies (row-major, transposed, and W)
  cvt_lin_kernel<<<4096, 256, 0, stream>>>(features, fmd, nfeat);
  cvt_tr_kernel <<<4096, 256, 0, stream>>>(features, fdm);
  cvt_lin_kernel<<<256, 256, 0, stream>>>(W, wbf, (size_t)D_ * D_);

  // Fused attention: one WG per (batch, 16-row query tile)
  dim3 grid(B_, N_ / 16);
  fused_attn_kernel<<<grid, 128, 0, stream>>>(lengths, bias, fmd, fdm, wbf, out);
}
